// InteractionLayer_14663018348663
// MI455X (gfx1250) — matrix operand order repaired and checked by reading, not compiled, not run on hardware
//
#include <hip/hip_runtime.h>
#include <hip/hip_bf16.h>
#include <math.h>

typedef __bf16 bf16;
typedef bf16 v16bf __attribute__((ext_vector_type(16)));
typedef bf16 v8bf  __attribute__((ext_vector_type(8)));
typedef float v8f  __attribute__((ext_vector_type(8)));

#define EPSF 1e-6f

__device__ __forceinline__ v8f wmma_bf16(v16bf a, v16bf b, v8f c) {
  // D = A(16x32 bf16) * B(32x16 bf16) + C(f32)
  return __builtin_amdgcn_wmma_f32_16x16x32_bf16(false, a, false, b, (short)0, c,
                                                 false, false);
}

__device__ __forceinline__ v16bf join8(v8bf lo, v8bf hi) {
  return __builtin_shufflevector(lo, hi, 0,1,2,3,4,5,6,7,8,9,10,11,12,13,14,15);
}

// A fragment (16x32 bf16) from an LDS tile, row-major [16][KLD].
// Lane l<16 holds row M=l, K {kb..kb+7, kb+16..kb+23}; lane l+16 holds K {+8,+24}.
template<int KLD>
__device__ __forceinline__ v16bf load_a_lds(const bf16* tile, int lane, int kbase) {
  int row = lane & 15, half = lane >> 4;
  const bf16* p = tile + row * KLD + kbase + half * 8;
  v8bf lo = *(const v8bf*)p;
  v8bf hi = *(const v8bf*)(p + 16);
  return join8(lo, hi);
}

// B fragment (32x16 bf16) from transposed weights WT[n][k] in global memory.
// Lane l<16 holds col n0+l, K {kb..kb+7, kb+16..kb+23}; lane l+16 holds K {+8,+24}.
__device__ __forceinline__ v16bf load_b_wt(const bf16* __restrict__ wt, int K,
                                           int n0, int kbase, int lane) {
  int col = lane & 15, half = lane >> 4;
  const bf16* p = wt + (size_t)(n0 + col) * K + kbase + half * 8;
  v8bf lo = *(const v8bf*)p;
  v8bf hi = *(const v8bf*)(p + 16);
  return join8(lo, hi);
}

// ---------------- prep: transpose f32 weights -> bf16 WT[n][k] ----------------
__global__ void k_prep(const float* __restrict__ src, bf16* __restrict__ dst,
                       int K, int Nn) {
  int i = blockIdx.x * 256 + threadIdx.x;
  if (i >= K * Nn) return;
  int n = i / K, k = i - n * K;
  dst[i] = (bf16)src[(size_t)k * Nn + n];
}

// ---------------- init node accumulators ----------------
__global__ void k_init(float* __restrict__ nden, float* __restrict__ envnum, int n32) {
  int i = blockIdx.x * 256 + threadIdx.x;
  if (i < n32) nden[i] = 0.0f;
  if (i < n32 * 4) envnum[i] = 0.0f;
}

// ---- edge phase 1: env weights, logits, fused scatter-softmax accumulation ----
// Note: logits are clipped to [-5,5] before softmax, so exp() needs no
// max-subtraction pass (range [6.7e-3, 148] is safe in f32 for any segment sum).
__global__ __launch_bounds__(128) void k_edge1(
    const float* __restrict__ node_attrs, const float* __restrict__ scalar_state,
    const float* __restrict__ cond, const int* __restrict__ esrc,
    const float* __restrict__ equiv,
    const bf16* __restrict__ W1T, const float* __restrict__ b1,
    const bf16* __restrict__ W2T, const bf16* __restrict__ WqT,
    const bf16* __restrict__ WkT,
    float* __restrict__ nden, float* __restrict__ envnum) {
  __shared__ __align__(16) bf16 xt[4][16][96];    // [scalar(64) | cond(32)] bf16
  __shared__ __align__(16) bf16 at[4][16][64];    // gathered node attrs
  __shared__ __align__(16) bf16 ht[4][16][128];   // silu hidden
  __shared__ __align__(16) float wt[4][16][64];   // per-edge env weights (f32)
  __shared__ __align__(16) float lt[4][16][32];   // per-edge logits (f32)

  const int wid = threadIdx.x >> 5;
  const int lane = threadIdx.x & 31;
  const int e0 = blockIdx.x * 64 + wid * 16;
  const int col = lane & 15, half = lane >> 4;
  const v8f vzero = {0.f,0.f,0.f,0.f,0.f,0.f,0.f,0.f};

  __builtin_prefetch(WqT, 0, 1);
  __builtin_prefetch(WkT, 0, 1);

  for (int idx = lane; idx < 16 * 96; idx += 32) {
    int r = idx / 96, c = idx - r * 96;
    float v = (c < 64) ? scalar_state[(size_t)(e0 + r) * 64 + c]
                       : cond[(size_t)(e0 + r) * 32 + (c - 64)];
    xt[wid][r][c] = (bf16)v;
  }
  for (int idx = lane; idx < 16 * 64; idx += 32) {
    int r = idx >> 6, c = idx & 63;
    int s = esrc[e0 + r];
    at[wid][r][c] = (bf16)node_attrs[(size_t)s * 64 + c];
  }
  __syncthreads();

  v16bf ax0 = load_a_lds<96>(&xt[wid][0][0], lane, 0);
  v16bf ax1 = load_a_lds<96>(&xt[wid][0][0], lane, 32);
  v16bf ax2 = load_a_lds<96>(&xt[wid][0][0], lane, 64);

  // h = silu(x @ W_env1 + b_env1)   (96 -> 128)
  #pragma unroll
  for (int nt = 0; nt < 8; nt++) {
    v8f acc = vzero;
    acc = wmma_bf16(ax0, load_b_wt(W1T, 96, nt * 16, 0, lane), acc);
    acc = wmma_bf16(ax1, load_b_wt(W1T, 96, nt * 16, 32, lane), acc);
    acc = wmma_bf16(ax2, load_b_wt(W1T, 96, nt * 16, 64, lane), acc);
    float bias = b1[nt * 16 + col];
    #pragma unroll
    for (int r = 0; r < 8; r++) {
      float v = acc[r] + bias;
      v = v / (1.0f + __expf(-v));
      ht[wid][half * 8 + r][nt * 16 + col] = (bf16)v;
    }
  }
  __syncthreads();

  v16bf ah0 = load_a_lds<128>(&ht[wid][0][0], lane, 0);
  v16bf ah1 = load_a_lds<128>(&ht[wid][0][0], lane, 32);
  v16bf ah2 = load_a_lds<128>(&ht[wid][0][0], lane, 64);
  v16bf ah3 = load_a_lds<128>(&ht[wid][0][0], lane, 96);

  // w = h @ W_env2   (128 -> 64), kept in LDS (f32)
  #pragma unroll
  for (int nt = 0; nt < 4; nt++) {
    v8f acc = vzero;
    acc = wmma_bf16(ah0, load_b_wt(W2T, 128, nt * 16, 0, lane), acc);
    acc = wmma_bf16(ah1, load_b_wt(W2T, 128, nt * 16, 32, lane), acc);
    acc = wmma_bf16(ah2, load_b_wt(W2T, 128, nt * 16, 64, lane), acc);
    acc = wmma_bf16(ah3, load_b_wt(W2T, 128, nt * 16, 96, lane), acc);
    #pragma unroll
    for (int r = 0; r < 8; r++)
      wt[wid][half * 8 + r][nt * 16 + col] = acc[r];
  }

  // logits[e,m] = clip( (a_e Wq_m) . (s_e Wk_m) / sqrt(16) ), kept in LDS
  v16bf aa0 = load_a_lds<64>(&at[wid][0][0], lane, 0);
  v16bf aa1 = load_a_lds<64>(&at[wid][0][0], lane, 32);

  for (int m = 0; m < 32; m++) {
    v8f q = vzero, kk = vzero;
    q  = wmma_bf16(aa0, load_b_wt(WqT, 64, m * 16, 0, lane), q);
    q  = wmma_bf16(aa1, load_b_wt(WqT, 64, m * 16, 32, lane), q);
    kk = wmma_bf16(ax0, load_b_wt(WkT, 64, m * 16, 0, lane), kk);
    kk = wmma_bf16(ax1, load_b_wt(WkT, 64, m * 16, 32, lane), kk);
    #pragma unroll
    for (int r = 0; r < 8; r++) {
      float p = q[r] * kk[r];
      p += __shfl_xor(p, 1);
      p += __shfl_xor(p, 2);
      p += __shfl_xor(p, 4);
      p += __shfl_xor(p, 8);
      float l = fminf(fmaxf(p * 0.25f, -5.0f), 5.0f);
      if (col == 0)
        lt[wid][half * 8 + r][m] = l;
    }
  }
  __syncthreads();

  // fused scatter-softmax numerators/denominators: m = lane (coalesced)
  const int m = lane;
  for (int r = 0; r < 16; r++) {
    int e = e0 + r;
    int s = esrc[e];
    float p = __expf(lt[wid][r][m]);
    float w0 = wt[wid][r][2 * m];
    float w1 = wt[wid][r][2 * m + 1];
    float4 eq = *(const float4*)&equiv[((size_t)e * 32 + m) * 4];
    atomicAdd(&nden[(size_t)s * 32 + m], p);
    float* dst = &envnum[((size_t)s * 32 + m) * 4];
    atomicAdd(dst + 0, p * eq.x * w0);
    atomicAdd(dst + 1, p * eq.y * w1);
    atomicAdd(dst + 2, p * eq.z * w1);
    atomicAdd(dst + 3, p * eq.w * w1);
  }
}

// ---------------- per-node SO(3) layernorm, blocks (1,3) ----------------
__global__ __launch_bounds__(256) void k_node(const float* __restrict__ envnum,
                                              const float* __restrict__ nden,
                                              const float* __restrict__ g_env,
                                              float* __restrict__ envnodes, int N) {
  int wid = threadIdx.x >> 5, lane = threadIdx.x & 31;
  int n = blockIdx.x * 8 + wid;
  if (n >= N) return;
  float4 num = *(const float4*)&envnum[((size_t)n * 32 + lane) * 4];
  float den = nden[(size_t)n * 32 + lane];
  float inv = den > 0.0f ? 1.0f / den : 0.0f;
  float e0 = num.x * inv, e1 = num.y * inv, e2 = num.z * inv, e3 = num.w * inv;
  float s0 = e0 * e0;
  float s1 = e1 * e1 + e2 * e2 + e3 * e3;
  #pragma unroll
  for (int mask = 1; mask < 32; mask <<= 1) {
    s0 += __shfl_xor(s0, mask);
    s1 += __shfl_xor(s1, mask);
  }
  float i0 = rsqrtf(s0 * (1.0f / 32.0f) + EPSF);
  float i1 = rsqrtf(s1 * (1.0f / 96.0f) + EPSF);
  float g0 = g_env[lane * 2], g1 = g_env[lane * 2 + 1];
  float4 o;
  o.x = e0 * i0 * g0;
  o.y = e1 * i1 * g1;
  o.z = e2 * i1 * g1;
  o.w = e3 * i1 * g1;
  *(float4*)&envnodes[((size_t)n * 32 + lane) * 4] = o;
}

// ---------------- edge phase 2: tensor product, norm, proj MLP, residual ----------------
__global__ __launch_bounds__(64) void k_edge2(
    const float* __restrict__ scalar_state, const float* __restrict__ equiv,
    const float* __restrict__ cond, const int* __restrict__ esrc,
    const float* __restrict__ envnodes, const float* __restrict__ g_tp,
    const bf16* __restrict__ Wp1T, const float* __restrict__ bp1,
    const bf16* __restrict__ Wp2T, const float* __restrict__ coeff,
    float* __restrict__ d_out, int E) {
  __shared__ __align__(16) bf16 tp[2][16][256];    // [row][m*8+j] normalized tp
  __shared__ __align__(16) float pwb[2][16][128];  // pw projection outputs
  __shared__ __align__(16) bf16 xp[2][16][96];
  __shared__ __align__(16) bf16 hp[2][16][128];
  __shared__ float ssum[2][16][4];
  __shared__ float invn[2][16][4];
  __shared__ int srcs[2][16];

  const int wid = threadIdx.x >> 5;
  const int lane = threadIdx.x & 31;
  const int e0 = blockIdx.x * 32 + wid * 16;
  const int col = lane & 15, half = lane >> 4;
  const v8f vzero = {0.f,0.f,0.f,0.f,0.f,0.f,0.f,0.f};

  __builtin_prefetch(Wp2T, 0, 1);

  float cv = coeff[0];
  float c_old = rsqrtf(cv * cv + 1.0f);
  float c_new = cv * c_old;

  if (lane < 16) {
    srcs[wid][lane] = esrc[e0 + lane];
    #pragma unroll
    for (int b = 0; b < 4; b++) ssum[wid][lane][b] = 0.0f;
  }
  for (int idx = lane; idx < 16 * 32; idx += 32) {
    int r = idx >> 5, c = idx & 31;
    xp[wid][r][64 + c] = (bf16)cond[(size_t)(e0 + r) * 32 + c];
  }
  __syncthreads();

  const float inv_s3 = 0.5773502691896258f;
  for (int idx = lane; idx < 512; idx += 32) {
    int m = idx >> 4, r = idx & 15;
    int e = e0 + r, s = srcs[wid][r];
    float4 eq = *(const float4*)&equiv[((size_t)e * 32 + m) * 4];
    float4 en = *(const float4*)&envnodes[((size_t)s * 32 + m) * 4];
    float o0a = eq.x * en.x;
    float o0b = (eq.y * en.y + eq.z * en.z + eq.w * en.w) * inv_s3;
    float a0 = eq.x * en.y, a1 = eq.x * en.z, a2 = eq.x * en.w;
    float b0 = eq.y * en.x, b1 = eq.z * en.x, b2 = eq.w * en.x;
    bf16* t = &tp[wid][r][m * 8];
    t[0] = (bf16)o0a; t[1] = (bf16)o0b;
    t[2] = (bf16)a0;  t[3] = (bf16)a1; t[4] = (bf16)a2;
    t[5] = (bf16)b0;  t[6] = (bf16)b1; t[7] = (bf16)b2;
    atomicAdd(&ssum[wid][r][0], o0a * o0a);
    atomicAdd(&ssum[wid][r][1], o0b * o0b);
    atomicAdd(&ssum[wid][r][2], a0 * a0 + a1 * a1 + a2 * a2);
    atomicAdd(&ssum[wid][r][3], b0 * b0 + b1 * b1 + b2 * b2);
  }
  __syncthreads();
  if (lane < 16) {
    invn[wid][lane][0] = rsqrtf(ssum[wid][lane][0] * (1.0f / 32.0f) + EPSF);
    invn[wid][lane][1] = rsqrtf(ssum[wid][lane][1] * (1.0f / 32.0f) + EPSF);
    invn[wid][lane][2] = rsqrtf(ssum[wid][lane][2] * (1.0f / 96.0f) + EPSF);
    invn[wid][lane][3] = rsqrtf(ssum[wid][lane][3] * (1.0f / 96.0f) + EPSF);
  }
  __syncthreads();

  // normalize tp, build xp scalars
  for (int idx = lane; idx < 512; idx += 32) {
    int m = idx >> 4, r = idx & 15;
    bf16* t = &tp[wid][r][m * 8];
    float i0 = invn[wid][r][0], i1 = invn[wid][r][1];
    float i2 = invn[wid][r][2], i3 = invn[wid][r][3];
    float g0 = g_tp[m * 4], g1 = g_tp[m * 4 + 1];
    float g2 = g_tp[m * 4 + 2], g3 = g_tp[m * 4 + 3];
    float v0 = (float)t[0] * i0 * g0;
    float v1 = (float)t[1] * i1 * g1;
    float v2 = (float)t[2] * i2 * g2, v3 = (float)t[3] * i2 * g2, v4 = (float)t[4] * i2 * g2;
    float v5 = (float)t[5] * i3 * g3, v6 = (float)t[6] * i3 * g3, v7 = (float)t[7] * i3 * g3;
    t[0] = (bf16)v0; t[1] = (bf16)v1; t[2] = (bf16)v2; t[3] = (bf16)v3;
    t[4] = (bf16)v4; t[5] = (bf16)v5; t[6] = (bf16)v6; t[7] = (bf16)v7;
    xp[wid][r][m * 2] = (bf16)v0;
    xp[wid][r][m * 2 + 1] = (bf16)v1;
  }
  __syncthreads();

  // hp = silu(xp @ Wp1 + bp1)   (96 -> 128)
  v16bf a0 = load_a_lds<96>(&xp[wid][0][0], lane, 0);
  v16bf a1 = load_a_lds<96>(&xp[wid][0][0], lane, 32);
  v16bf a2 = load_a_lds<96>(&xp[wid][0][0], lane, 64);
  #pragma unroll
  for (int nt = 0; nt < 8; nt++) {
    v8f acc = vzero;
    acc = wmma_bf16(a0, load_b_wt(Wp1T, 96, nt * 16, 0, lane), acc);
    acc = wmma_bf16(a1, load_b_wt(Wp1T, 96, nt * 16, 32, lane), acc);
    acc = wmma_bf16(a2, load_b_wt(Wp1T, 96, nt * 16, 64, lane), acc);
    float bias = bp1[nt * 16 + col];
    #pragma unroll
    for (int r = 0; r < 8; r++) {
      float v = acc[r] + bias;
      v = v / (1.0f + __expf(-v));
      hp[wid][half * 8 + r][nt * 16 + col] = (bf16)v;
    }
  }
  __syncthreads();

  // op = hp @ Wp2   (128 -> 192): cols 0..63 -> scalar residual, 64..191 -> pw
  v16bf h0 = load_a_lds<128>(&hp[wid][0][0], lane, 0);
  v16bf h1 = load_a_lds<128>(&hp[wid][0][0], lane, 32);
  v16bf h2 = load_a_lds<128>(&hp[wid][0][0], lane, 64);
  v16bf h3 = load_a_lds<128>(&hp[wid][0][0], lane, 96);
  #pragma unroll
  for (int nt = 0; nt < 12; nt++) {
    v8f acc = vzero;
    acc = wmma_bf16(h0, load_b_wt(Wp2T, 128, nt * 16, 0, lane), acc);
    acc = wmma_bf16(h1, load_b_wt(Wp2T, 128, nt * 16, 32, lane), acc);
    acc = wmma_bf16(h2, load_b_wt(Wp2T, 128, nt * 16, 64, lane), acc);
    acc = wmma_bf16(h3, load_b_wt(Wp2T, 128, nt * 16, 96, lane), acc);
    if (nt < 4) {
      int gcol = nt * 16 + col;
      #pragma unroll
      for (int r = 0; r < 8; r++) {
        int e = e0 + half * 8 + r;
        d_out[(size_t)e * 64 + gcol] =
            c_old * scalar_state[(size_t)e * 64 + gcol] + c_new * acc[r];
      }
    } else {
      int pcol = nt * 16 + col - 64;
      #pragma unroll
      for (int r = 0; r < 8; r++) pwb[wid][half * 8 + r][pcol] = acc[r];
    }
  }
  __syncthreads();

  // new_equiv + residual
  size_t eqout = (size_t)E * 64;
  for (int idx = lane; idx < 512; idx += 32) {
    int m = idx >> 4, r = idx & 15;
    int e = e0 + r;
    const bf16* t = &tp[wid][r][m * 8];
    float p0 = pwb[wid][r][m * 4 + 0], p1 = pwb[wid][r][m * 4 + 1];
    float p2 = pwb[wid][r][m * 4 + 2], p3 = pwb[wid][r][m * 4 + 3];
    float4 eq = *(const float4*)&equiv[((size_t)e * 32 + m) * 4];
    float ns  = p0 * (float)t[0] + p1 * (float)t[1];
    float nv0 = p2 * (float)t[2] + p3 * (float)t[5];
    float nv1 = p2 * (float)t[3] + p3 * (float)t[6];
    float nv2 = p2 * (float)t[4] + p3 * (float)t[7];
    float4 o;
    o.x = c_old * eq.x + c_new * ns;
    o.y = c_old * eq.y + c_new * nv0;
    o.z = c_old * eq.z + c_new * nv1;
    o.w = c_old * eq.w + c_new * nv2;
    *(float4*)&d_out[eqout + ((size_t)e * 32 + m) * 4] = o;
  }
}

extern "C" void kernel_launch(void* const* d_in, const int* in_sizes, int n_in,
                              void* d_out, int out_size, void* d_ws, size_t ws_size,
                              hipStream_t stream) {
  const float* node_attrs = (const float*)d_in[0];
  const float* scalar_state = (const float*)d_in[1];
  const float* equiv = (const float*)d_in[2];
  const float* cond = (const float*)d_in[3];
  const float* coeff = (const float*)d_in[4];
  const float* W_env1 = (const float*)d_in[5];
  const float* b_env1 = (const float*)d_in[6];
  const float* W_env2 = (const float*)d_in[7];
  const float* Wq = (const float*)d_in[8];
  const float* Wk = (const float*)d_in[9];
  const float* g_env = (const float*)d_in[10];
  const float* g_tp = (const float*)d_in[11];
  const float* Wp1 = (const float*)d_in[12];
  const float* bp1 = (const float*)d_in[13];
  const float* Wp2 = (const float*)d_in[14];
  const int* esrc = (const int*)d_in[15];

  const int N = in_sizes[0] / 64;
  const int E = in_sizes[1] / 64;
  float* out = (float*)d_out;

  // workspace layout: bf16 transposed weights, then f32 node accumulators
  char* ws = (char*)d_ws;
  bf16* W1T = (bf16*)ws;            // 128 x 96
  bf16* W2T = W1T + 128 * 96;       // 64  x 128
  bf16* WqT = W2T + 64 * 128;       // 512 x 64
  bf16* WkT = WqT + 512 * 64;       // 512 x 64
  bf16* Wp1T = WkT + 512 * 64;      // 128 x 96
  bf16* Wp2T = Wp1T + 128 * 96;     // 192 x 128
  size_t wend = (size_t)(128*96 + 64*128 + 512*64 + 512*64 + 128*96 + 192*128) * 2;
  float* fbase = (float*)(ws + wend);            // wend = 245760, 16B aligned
  float* d_nden = fbase;                         // N*32
  float* d_envnum = d_nden + (size_t)N * 32;     // N*128
  float* d_envnodes = d_envnum + (size_t)N * 128; // N*128

  // weight transposes (f32 -> bf16 [n][k])
  k_prep<<<(96 * 128 + 255) / 256, 256, 0, stream>>>(W_env1, W1T, 96, 128);
  k_prep<<<(128 * 64 + 255) / 256, 256, 0, stream>>>(W_env2, W2T, 128, 64);
  k_prep<<<(64 * 512 + 255) / 256, 256, 0, stream>>>(Wq, WqT, 64, 512);
  k_prep<<<(64 * 512 + 255) / 256, 256, 0, stream>>>(Wk, WkT, 64, 512);
  k_prep<<<(96 * 128 + 255) / 256, 256, 0, stream>>>(Wp1, Wp1T, 96, 128);
  k_prep<<<(128 * 192 + 255) / 256, 256, 0, stream>>>(Wp2, Wp2T, 128, 192);

  k_init<<<(N * 128 + 255) / 256, 256, 0, stream>>>(d_nden, d_envnum, N * 32);

  k_edge1<<<E / 64, 128, 0, stream>>>(node_attrs, scalar_state, cond, esrc, equiv,
                                      W1T, b_env1, W2T, WqT, WkT, d_nden, d_envnum);

  k_node<<<(N + 7) / 8, 256, 0, stream>>>(d_envnum, d_nden, g_env, d_envnodes, N);

  k_edge2<<<E / 32, 64, 0, stream>>>(scalar_state, equiv, cond, esrc, d_envnodes,
                                     g_tp, Wp1T, bp1, Wp2T, coeff, out, E);
}
